// SMRNN_64613488001380
// MI455X (gfx1250) — compile-verified, matrix operand order checked by reading, not run on hardware
//
#include <hip/hip_runtime.h>
#include <math.h>

typedef __attribute__((ext_vector_type(16))) _Float16 v16h;
typedef __attribute__((ext_vector_type(8)))  float    v8f;
typedef __attribute__((ext_vector_type(4)))  float    v4f;

/* explicit LDS (addrspace 3) vector pointer => guaranteed ds_load_b128 */
typedef __attribute__((address_space(3))) const v4f* lds_v4f_cptr;

#define D_IN     640
#define W_TOTAL  802816
#define B_TOTAL  1728
#define NROWS    (W_TOTAL + B_TOTAL)   /* 804544 */
#define NGROUPS  (NROWS / 16)          /* 50284  */

/* LDS tile: 16 rows x 64 floats, padded row stride 68 floats (272B) for
   bank-conflict-free ds_load_b128 reads. */
#define TILE_ROW_F   68
#define TILE_F       (16 * TILE_ROW_F)   /* 1088 floats = 4352 B */
#define TILE_B       (TILE_F * 4)

/* workspace layout (float element offsets) */
#define WS_INPT  0
#define WS_H0    640
#define WS_H1    1280
#define WS_H2    1792
#define WS_HN    1856
#define WS_G1    2368
#define WS_G2    3008
#define WS_G3    3520
#define WS_X2    4096   /* 804544 floats */

// ---------------------------------------------------------------------------
// concat inpt = [x(64) | prev_output(64) | state(512)]
// ---------------------------------------------------------------------------
__global__ void smrnn_concat(const float* __restrict__ x,
                             const float* __restrict__ prev,
                             const float* __restrict__ st,
                             float* __restrict__ inpt) {
    int i = blockIdx.x * blockDim.x + threadIdx.x;
    if (i < 64)        inpt[i] = x[i];
    else if (i < 128)  inpt[i] = prev[i - 64];
    else if (i < 640)  inpt[i] = st[i - 128];
}

// ---------------------------------------------------------------------------
// generic fused GEMV: y = act(W @ x + b); W row-major (out_dim, in_dim)
// act: 0 = none, 1 = leaky(0.01)
// ---------------------------------------------------------------------------
__global__ void smrnn_gemv(const float* __restrict__ W,
                           const float* __restrict__ b,
                           const float* __restrict__ x,
                           float* __restrict__ y,
                           int in_dim, int out_dim, int act) {
    __shared__ __align__(16) float sx[D_IN];
    for (int i = threadIdx.x; i < in_dim; i += blockDim.x) sx[i] = x[i];
    __syncthreads();

    int r = blockIdx.x * blockDim.x + threadIdx.x;
    if (r >= out_dim) return;

    const float4* wr = (const float4*)(W + (size_t)r * in_dim);
    const float4* xv = (const float4*)sx;
    int n4 = in_dim >> 2;
    float a0 = 0.f, a1 = 0.f, a2 = 0.f, a3 = 0.f;
#pragma unroll 4
    for (int i = 0; i < n4; ++i) {
        float4 w = wr[i];
        float4 v = xv[i];
        a0 = fmaf(w.x, v.x, a0);
        a1 = fmaf(w.y, v.y, a1);
        a2 = fmaf(w.z, v.z, a2);
        a3 = fmaf(w.w, v.w, a3);
    }
    float acc = (a0 + a1) + (a2 + a3) + b[r];
    if (act == 1) acc = (acc >= 0.f) ? acc : 0.01f * acc;
    y[r] = acc;
}

// ---------------------------------------------------------------------------
// Async global->LDS helpers (CDNA5 ASYNCcnt path)
// ---------------------------------------------------------------------------
__device__ __forceinline__ unsigned lds_off(const float* p) {
    // generic -> LDS(as3) pointer -> 32-bit LDS byte address
    return (unsigned)(unsigned long long)(__attribute__((address_space(3))) const float*)p;
}

__device__ __forceinline__ void async_copy16(unsigned lds_byte, const float* gptr) {
    asm volatile("global_load_async_to_lds_b128 %0, %1, off"
                 :: "v"(lds_byte), "v"(gptr));
}

__device__ __forceinline__ void wait_async_le8() {
    asm volatile("s_wait_asynccnt 0x8" ::: "memory");
}
__device__ __forceinline__ void wait_async_0() {
    asm volatile("s_wait_asynccnt 0x0" ::: "memory");
}

// Issue one 16x64 f32 tile (4KB) as 8 async b128 loads per lane.
// Global bytes [i*512 + lane*16); LDS row-padded: row r -> r*272B.
__device__ __forceinline__ void issue_tile(const float* __restrict__ W3, int g,
                                           unsigned lbase_b, int lane) {
    const float* gbase = W3 + (size_t)g * 1024;          // 16 rows * 64 floats
    const int chunk = lane & 15;
    const int rpar  = lane >> 4;
#pragma unroll
    for (int i = 0; i < 8; ++i) {
        int r = i * 2 + rpar;
        unsigned la = lbase_b + (unsigned)(r * 272 + chunk * 16);
        const float* gp = gbase + (size_t)i * 128 + (size_t)lane * 4;
        async_copy16(la, gp);
    }
}

// ---------------------------------------------------------------------------
// Big hyper-weight GEMV: x2 = tanh(W3 @ h + b3) * 0.5, W3 is (804544 x 64).
// Per wave: double-buffered async LDS tiles; 16 rows x 64 k per iteration via
// two v_wmma_f32_16x16x32_f16. B = h replicated across all 16 columns, so
// every column of D holds the same 16 dot products (trivial extraction).
// ---------------------------------------------------------------------------
__device__ __forceinline__ void put4(v16h& a, int e, v4f w) {
    a[e + 0] = (_Float16)w[0];
    a[e + 1] = (_Float16)w[1];
    a[e + 2] = (_Float16)w[2];
    a[e + 3] = (_Float16)w[3];
}

__global__ void __launch_bounds__(256)
smrnn_hyper_wmma(const float* __restrict__ W3,
                 const float* __restrict__ B3,
                 const float* __restrict__ hsrc,
                 float* __restrict__ x2) {
    __shared__ __align__(16) float tiles[8 * 2 * TILE_F];   // 8 waves x 2 bufs
    __shared__ __align__(16) float sh[64];
    if (threadIdx.x < 64) sh[threadIdx.x] = hsrc[threadIdx.x];
    __syncthreads();

    const int lane   = threadIdx.x & 31;
    const int wslot  = threadIdx.x >> 5;
    const int wave   = (blockIdx.x * blockDim.x + threadIdx.x) >> 5;
    const int nwaves = (gridDim.x * blockDim.x) >> 5;
    // 16-bit A/B VGPR layout: lanes 0-15 own k {0..7,16..23}, lanes 16-31 own
    // k {8..15,24..31} (shifted by +8 cols). cb = column base for this lane.
    const int cb = (lane < 16) ? 0 : 8;

    // LDS byte addresses only (no generic pointers in the hot loop).
    const unsigned wbase_b = lds_off(tiles) + (unsigned)(wslot * 2 * TILE_B);
    const unsigned rd_lane_b = (unsigned)(((lane & 15) * TILE_ROW_F + cb) * 4);

    // Build replicated-column B tiles for k=[0,32) and k=[32,64) once.
    v16h b0, b1;
#pragma unroll
    for (int e = 0; e < 8; ++e) {
        b0[e]     = (_Float16)sh[cb + e];
        b0[e + 8] = (_Float16)sh[cb + 16 + e];
        b1[e]     = (_Float16)sh[cb + 32 + e];
        b1[e + 8] = (_Float16)sh[cb + 48 + e];
    }

    int g = wave;
    unsigned curoff = 0;
    if (g < NGROUPS) issue_tile(W3, g, wbase_b, lane);

    for (; g < NGROUPS; g += nwaves) {
        const int gn = g + nwaves;                 // wave-uniform
        if (gn < NGROUPS) {
            issue_tile(W3, gn, wbase_b + (curoff ^ TILE_B), lane);
            wait_async_le8();                      // oldest 8 (current tile) done
        } else {
            wait_async_0();
        }

        // Conflict-free ds_load_b128 reads via explicit as(3) pointer.
        lds_v4f_cptr p = (lds_v4f_cptr)(wbase_b + curoff + rd_lane_b);
        v4f w0 = p[0],  w1 = p[1];
        v4f w2 = p[4],  w3 = p[5];
        v4f w4 = p[8],  w5 = p[9];
        v4f w6 = p[12], w7 = p[13];

        v16h a0, a1;
        put4(a0, 0, w0);  put4(a0, 4, w1);  put4(a0, 8, w2);  put4(a0, 12, w3);
        put4(a1, 0, w4);  put4(a1, 4, w5);  put4(a1, 8, w6);  put4(a1, 12, w7);

        v8f c = {};
        c = __builtin_amdgcn_wmma_f32_16x16x32_f16(false, a0, false, b0,
                                                   (short)0, c, false, false);
        c = __builtin_amdgcn_wmma_f32_16x16x32_f16(false, a1, false, b1,
                                                   (short)0, c, false, false);

        // lanes 0/16 hold rows r0..r0+7 / r0+8..r0+15 in c[0..7]
        if ((lane & 15) == 0) {
            const int m0 = (g << 4) + ((lane >> 4) << 3);
#pragma unroll
            for (int j = 0; j < 8; ++j) {
                float v = c[j] + B3[m0 + j];
                x2[m0 + j] = tanhf(v) * 0.5f;
            }
        }
        curoff ^= TILE_B;
    }
}

// ---------------------------------------------------------------------------
extern "C" void kernel_launch(void* const* d_in, const int* in_sizes, int n_in,
                              void* d_out, int out_size, void* d_ws, size_t ws_size,
                              hipStream_t stream) {
    const float* x       = (const float*)d_in[0];
    const float* prev    = (const float*)d_in[1];
    const float* st      = (const float*)d_in[2];
    const float* n1_w0   = (const float*)d_in[3];
    const float* n1_b0   = (const float*)d_in[4];
    const float* n1_w1   = (const float*)d_in[5];
    const float* n1_b1   = (const float*)d_in[6];
    const float* n2_w0   = (const float*)d_in[7];
    const float* n2_b0   = (const float*)d_in[8];
    const float* n2_w1   = (const float*)d_in[9];
    const float* n2_b1   = (const float*)d_in[10];
    const float* n2_w2   = (const float*)d_in[11];
    const float* n2_b2   = (const float*)d_in[12];
    const float* n2_w3   = (const float*)d_in[13];
    const float* n2_b3   = (const float*)d_in[14];

    float* ws   = (float*)d_ws;
    float* inpt = ws + WS_INPT;
    float* h0   = ws + WS_H0;
    float* h1   = ws + WS_H1;
    float* h2   = ws + WS_H2;
    float* hn   = ws + WS_HN;
    float* g1   = ws + WS_G1;
    float* g2   = ws + WS_G2;
    float* g3   = ws + WS_G3;
    float* x2   = ws + WS_X2;
    float* out  = (float*)d_out;

    dim3 blk(256);

    // inpt = concat(x, prev, state)
    smrnn_concat<<<dim3(3), blk, 0, stream>>>(x, prev, st, inpt);

    // net2 trunk
    smrnn_gemv<<<dim3(3), blk, 0, stream>>>(n2_w0, n2_b0, inpt, h0, 640, 640, 1);
    smrnn_gemv<<<dim3(2), blk, 0, stream>>>(n2_w1, n2_b1, h0,   h1, 640, 512, 1);
    smrnn_gemv<<<dim3(1), blk, 0, stream>>>(n2_w2, n2_b2, h1,   h2, 512, 64,  1);

    // big hyper-weight generation (HBM-bound, async LDS pipeline + WMMA)
    smrnn_hyper_wmma<<<dim3(2048), blk, 0, stream>>>(n2_w3, n2_b3, h2, x2);

    // generated net3: 640 -> 640 -> 512 -> 64 -> 512
    smrnn_gemv<<<dim3(3), blk, 0, stream>>>(x2 + 0,      x2 + W_TOTAL + 0,    inpt, g1, 640, 640, 1);
    smrnn_gemv<<<dim3(2), blk, 0, stream>>>(x2 + 409600, x2 + W_TOTAL + 640,  g1,   g2, 640, 512, 1);
    smrnn_gemv<<<dim3(1), blk, 0, stream>>>(x2 + 737280, x2 + W_TOTAL + 1152, g2,   g3, 512, 64,  1);
    smrnn_gemv<<<dim3(2), blk, 0, stream>>>(x2 + 770048, x2 + W_TOTAL + 1216, g3, out + 64, 64, 512, 0);

    // net1 (independent path)
    smrnn_gemv<<<dim3(2), blk, 0, stream>>>(n1_w0, n1_b0, inpt, hn, 640, 512, 1);
    smrnn_gemv<<<dim3(1), blk, 0, stream>>>(n1_w1, n1_b1, hn, out, 512, 64, 0);
}
